// TemporalSelfAttention_12128987644295
// MI455X (gfx1250) — matrix operand order repaired and checked by reading, not compiled, not run on hardware
//
#include <hip/hip_runtime.h>

// ---------------------------------------------------------------------------
// Fused temporal self-attention for MI455X (gfx1250, wave32, WMMA).
// One workgroup (8 wave32) per (b, n); wave w owns the 16-row stripe
// rows [16w, 16w+16) for every GEMM. All matmuls via
// v_wmma_f32_16x16x32_f16 (f32 accumulate). X A-fragments cached in VGPRs
// for the whole kernel; output accumulator lives in VGPRs across heads.
// Weights pre-converted to f16 + transposed in d_ws so B-fragments are
// pairs of 16B loads per lane (L2-resident across the 400 blocks).
// ---------------------------------------------------------------------------

typedef __attribute__((ext_vector_type(16))) _Float16 v16h;
typedef __attribute__((ext_vector_type(8)))  _Float16 v8h;
typedef __attribute__((ext_vector_type(4)))  _Float16 v4h;
typedef __attribute__((ext_vector_type(8)))  float    v8f;

#define B_DIM 4
#define T_DIM 128
#define N_DIM 100
#define D_MODEL 256
#define H_DIM 8
#define DK 128
#define INNER 1024

#define XP 264   // x staging pitch (halves): 256 + 8 pad
#define QP 136   // Q/K/Vt/P pitch (halves): 128 + 8 pad

// Xh (phase 0 only) overlays the Qh/Kh/Vt region: max(128*XP, 3*128*QP) halves.
#define SMEM_BYTES (3 * T_DIM * QP * 2)   // 104448 > 128*XP*2 = 67584

#define WMMA_F16(a, b, c) \
  __builtin_amdgcn_wmma_f32_16x16x32_f16(false, (a), false, (b), (short)0, (c), false, false)

// A-fragment (16x32 f16). ISA: lanes 0-15 row M=lane hold K = kBase+{0..7,16..23};
// lanes 16-31 row M=lane-16 hold K = kBase+{8..15, 24..31}.
static __device__ __forceinline__ v16h load_frag_a(const _Float16* buf, int pitch,
                                                   int rowBase, int kBase, int lane) {
  int row = rowBase + (lane & 15);
  int kb  = kBase + ((lane >> 4) << 3);
  const _Float16* p = buf + row * pitch + kb;
  v8h lo = *(const v8h*)(p);
  v8h hi = *(const v8h*)(p + 16);
  return __builtin_shufflevector(lo, hi, 0,1,2,3,4,5,6,7,8,9,10,11,12,13,14,15);
}

// B-fragment (32x16 f16), buffer stored [column][k] contiguous in k.
// Lanes 0-15 column=lane hold K = kBase+{0..7,16..23}; lanes 16-31 hold K = kBase+{8..15,24..31}.
static __device__ __forceinline__ v16h load_frag_b(const _Float16* buf, int pitch,
                                                   int colBase, int kBase, int lane) {
  int col = colBase + (lane & 15);
  int kb  = kBase + ((lane >> 4) << 3);
  const _Float16* p = buf + col * pitch + kb;
  v8h lo = *(const v8h*)(p);
  v8h hi = *(const v8h*)(p + 16);
  return __builtin_shufflevector(lo, hi, 0,1,2,3,4,5,6,7,8,9,10,11,12,13,14,15);
}

// ---------------------------------------------------------------------------
// Prep: f32 -> f16 transposed weights.
//   WqT/WkT/WvT : [INNER][D_MODEL]  (row e, col d)  from w[d*INNER + e]
//   WoT         : [D_MODEL][INNER]  (row d, col e)  from wo[e*D_MODEL + d]
// ---------------------------------------------------------------------------
__global__ void prep_weights_kernel(const float* __restrict__ wq, const float* __restrict__ wk,
                                    const float* __restrict__ wv, const float* __restrict__ wo,
                                    _Float16* __restrict__ WqT, _Float16* __restrict__ WkT,
                                    _Float16* __restrict__ WvT, _Float16* __restrict__ WoT) {
  int idx = blockIdx.x * 256 + threadIdx.x;      // 4 * 262144 total
  int m = idx >> 18;
  int r = idx & 262143;
  if (m < 3) {
    const float* src = (m == 0) ? wq : (m == 1) ? wk : wv;
    _Float16*    dst = (m == 0) ? WqT : (m == 1) ? WkT : WvT;
    int e = r >> 8, d = r & 255;
    dst[e * D_MODEL + d] = (_Float16)src[d * INNER + e];
  } else {
    int d = r >> 10, e = r & 1023;
    WoT[d * INNER + e] = (_Float16)wo[e * D_MODEL + d];
  }
}

// ---------------------------------------------------------------------------
// Main fused kernel: one block per (b, n).
// ---------------------------------------------------------------------------
__global__ __launch_bounds__(256, 1)
void attn_fused_kernel(const float* __restrict__ x,
                       const float* __restrict__ bq, const float* __restrict__ bk,
                       const float* __restrict__ bv, const float* __restrict__ bo,
                       const float* __restrict__ gamma, const float* __restrict__ beta,
                       const _Float16* __restrict__ WqT, const _Float16* __restrict__ WkT,
                       const _Float16* __restrict__ WvT, const _Float16* __restrict__ WoT,
                       float* __restrict__ out) {
  extern __shared__ char smem[];
  _Float16* Xh = (_Float16*)smem;                 // [128][XP]   (phase 0 only)
  _Float16* Qh = (_Float16*)smem;                 // [128][QP]   (Q, then P, then O staging)
  _Float16* Kh = Qh + T_DIM * QP;                 // [128][QP]
  _Float16* Vt = Kh + T_DIM * QP;                 // [128][QP]   V transposed: [dk][t]

  const int b = blockIdx.x / N_DIM;
  const int n = blockIdx.x % N_DIM;
  const int tid = threadIdx.x;
  const int wave = tid >> 5;
  const int lane = tid & 31;
  const int lane15 = lane & 15;
  const int rowOff8 = (lane >> 4) << 3;           // D-tile: vgpr r -> row r or r+8
  const int rowT = wave << 4;                     // this wave's 16-row stripe

  // ---- Phase 0: stage x[b, :, n, :] into LDS as f16 -----------------------
  for (int i = tid; i < T_DIM * (D_MODEL / 4); i += 256) {
    int row = i >> 6;
    int c4  = (i & 63) << 2;
    const float4 v = *(const float4*)(x + (((size_t)b * T_DIM + row) * N_DIM + n) * D_MODEL + c4);
    v4h hh;
    hh[0] = (_Float16)v.x; hh[1] = (_Float16)v.y; hh[2] = (_Float16)v.z; hh[3] = (_Float16)v.w;
    *(v4h*)(Xh + row * XP + c4) = hh;
  }
  __syncthreads();

  // ---- Cache X A-fragments for this stripe (reused by every projection) ---
  v16h xa[8];
#pragma unroll
  for (int kk = 0; kk < 8; ++kk) xa[kk] = load_frag_a(Xh, XP, rowT, kk * 32, lane);
  __syncthreads();                                // Xh region may now be overwritten

  const float qscale = 0.08838834764831845f;      // 1/sqrt(128)
  v8f acc[16];                                    // out-proj accumulator [16 rows][256 cols]
#pragma unroll
  for (int ct = 0; ct < 16; ++ct) acc[ct] = (v8f){};

  for (int h = 0; h < H_DIM; ++h) {
    if (h + 1 < H_DIM) {                          // warm L2/WGP$ for next head's weights
      const int nb = (h + 1) * DK * D_MODEL + tid * 128;
      __builtin_prefetch(WqT + nb, 0, 0);
      __builtin_prefetch(WkT + nb, 0, 0);
      __builtin_prefetch(WvT + nb, 0, 0);
    }

    // ---- Projections: wave w computes its 16-row stripe of Q, K, V --------
#pragma unroll
    for (int mat = 0; mat < 3; ++mat) {
      const _Float16* WT  = (mat == 0) ? WqT : (mat == 1) ? WkT : WvT;
      const float*    bia = (mat == 0) ? bq  : (mat == 1) ? bk  : bv;
      const float     sc  = (mat == 0) ? qscale : 1.0f;
      for (int ct = 0; ct < 8; ++ct) {
        v16h bfr[8];
#pragma unroll
        for (int kk = 0; kk < 8; ++kk)
          bfr[kk] = load_frag_b(WT, D_MODEL, h * DK + (ct << 4), kk * 32, lane);
        v8f c = {};
#pragma unroll
        for (int kk = 0; kk < 8; ++kk) c = WMMA_F16(xa[kk], bfr[kk], c);
        const float bval = bia[h * DK + (ct << 4) + lane15];
        if (mat != 2) {
          _Float16* dst = (mat == 0) ? Qh : Kh;
#pragma unroll
          for (int r = 0; r < 8; ++r)
            dst[(rowT + rowOff8 + r) * QP + (ct << 4) + lane15] = (_Float16)((c[r] + bval) * sc);
        } else {
          v8h pk;
#pragma unroll
          for (int r = 0; r < 8; ++r) pk[r] = (_Float16)(c[r] + bval);
          *(v8h*)(Vt + ((ct << 4) + lane15) * QP + rowT + rowOff8) = pk;
        }
      }
    }
    __syncthreads();

    // ---- S = Q K^T for this stripe, softmax fully in registers ------------
    v16h qa[4];
#pragma unroll
    for (int kk = 0; kk < 4; ++kk) qa[kk] = load_frag_a(Qh, QP, rowT, kk * 32, lane);

    v8f S[8];
#pragma unroll
    for (int nt = 0; nt < 8; ++nt) {
      v16h bfr[4];
#pragma unroll
      for (int kk = 0; kk < 4; ++kk)
        bfr[kk] = load_frag_b(Kh, QP, nt << 4, kk * 32, lane);
      v8f c = {};
#pragma unroll
      for (int kk = 0; kk < 4; ++kk) c = WMMA_F16(qa[kk], bfr[kk], c);
      S[nt] = c;
    }

#pragma unroll
    for (int r = 0; r < 8; ++r) {
      float m = S[0][r];
#pragma unroll
      for (int nt = 1; nt < 8; ++nt) m = fmaxf(m, S[nt][r]);
      m = fmaxf(m, __shfl_xor(m, 1, 16));
      m = fmaxf(m, __shfl_xor(m, 2, 16));
      m = fmaxf(m, __shfl_xor(m, 4, 16));
      m = fmaxf(m, __shfl_xor(m, 8, 16));
      float s = 0.0f;
#pragma unroll
      for (int nt = 0; nt < 8; ++nt) {
        float e = __expf(S[nt][r] - m);
        S[nt][r] = e;
        s += e;
      }
      s += __shfl_xor(s, 1, 16);
      s += __shfl_xor(s, 2, 16);
      s += __shfl_xor(s, 4, 16);
      s += __shfl_xor(s, 8, 16);
      const float inv = 1.0f / s;
#pragma unroll
      for (int nt = 0; nt < 8; ++nt) S[nt][r] *= inv;
    }

    // P -> LDS (own stripe only; aliases Qh, safe: qa already consumed)
#pragma unroll
    for (int nt = 0; nt < 8; ++nt)
#pragma unroll
      for (int r = 0; r < 8; ++r)
        Qh[(rowT + rowOff8 + r) * QP + (nt << 4) + lane15] = (_Float16)S[nt][r];

    v16h pa[4];
#pragma unroll
    for (int kk = 0; kk < 4; ++kk) pa[kk] = load_frag_a(Qh, QP, rowT, kk * 32, lane);

    // ---- O = P V  (B from Vt: [dk][t], contiguous in t) -------------------
    v8f O[8];
#pragma unroll
    for (int nt = 0; nt < 8; ++nt) {
      v16h bfr[4];
#pragma unroll
      for (int kk = 0; kk < 4; ++kk)
        bfr[kk] = load_frag_b(Vt, QP, nt << 4, kk * 32, lane);
      v8f c = {};
#pragma unroll
      for (int kk = 0; kk < 4; ++kk) c = WMMA_F16(pa[kk], bfr[kk], c);
      O[nt] = c;
    }

    // O -> LDS f16 (own stripe; pa consumed), reload in A layout
#pragma unroll
    for (int nt = 0; nt < 8; ++nt)
#pragma unroll
      for (int r = 0; r < 8; ++r)
        Qh[(rowT + rowOff8 + r) * QP + (nt << 4) + lane15] = (_Float16)O[nt][r];

    v16h oa[4];
#pragma unroll
    for (int kk = 0; kk < 4; ++kk) oa[kk] = load_frag_a(Qh, QP, rowT, kk * 32, lane);

    // ---- acc += O_h * Wo[h*128 : h*128+128, :]  (register accumulate) -----
#pragma unroll
    for (int ct = 0; ct < 16; ++ct) {
      v16h bfr[4];
#pragma unroll
      for (int kk = 0; kk < 4; ++kk)
        bfr[kk] = load_frag_b(WoT, INNER, ct << 4, h * DK + kk * 32, lane);
      v8f c = acc[ct];
#pragma unroll
      for (int kk = 0; kk < 4; ++kk) c = WMMA_F16(oa[kk], bfr[kk], c);
      acc[ct] = c;
    }
    __syncthreads();                              // protect Qh/Kh/Vt for next head
  }

  // ---- Epilogue: +bo, +residual x, LayerNorm(D=256), store f32 ------------
  float g16[16], bt16[16], bo16[16];
#pragma unroll
  for (int ct = 0; ct < 16; ++ct) {
    const int col = (ct << 4) + lane15;
    g16[ct]  = gamma[col];
    bt16[ct] = beta[col];
    bo16[ct] = bo[col];
  }

#pragma unroll
  for (int r = 0; r < 8; ++r) {
    const int row = rowT + rowOff8 + r;           // lanes 0-15: row rowT+r; 16-31: +8
    const size_t gbase = (((size_t)b * T_DIM + row) * N_DIM + n) * D_MODEL;
    float val[16];
    float s = 0.0f;
#pragma unroll
    for (int ct = 0; ct < 16; ++ct) {
      const int col = (ct << 4) + lane15;
      val[ct] = acc[ct][r] + bo16[ct] + x[gbase + col];
      s += val[ct];
    }
    s += __shfl_xor(s, 1, 16);
    s += __shfl_xor(s, 2, 16);
    s += __shfl_xor(s, 4, 16);
    s += __shfl_xor(s, 8, 16);
    const float mu = s * (1.0f / 256.0f);
    float v2 = 0.0f;
#pragma unroll
    for (int ct = 0; ct < 16; ++ct) {
      const float dlt = val[ct] - mu;
      v2 += dlt * dlt;
    }
    v2 += __shfl_xor(v2, 1, 16);
    v2 += __shfl_xor(v2, 2, 16);
    v2 += __shfl_xor(v2, 4, 16);
    v2 += __shfl_xor(v2, 8, 16);
    const float rstd = rsqrtf(v2 * (1.0f / 256.0f) + 1e-5f);
#pragma unroll
    for (int ct = 0; ct < 16; ++ct) {
      const int col = (ct << 4) + lane15;
      out[gbase + col] = (val[ct] - mu) * rstd * g16[ct] + bt16[ct];
    }
  }
}

// ---------------------------------------------------------------------------
extern "C" void kernel_launch(void* const* d_in, const int* in_sizes, int n_in,
                              void* d_out, int out_size, void* d_ws, size_t ws_size,
                              hipStream_t stream) {
  const float* x     = (const float*)d_in[0];
  const float* wq    = (const float*)d_in[1];
  const float* bq    = (const float*)d_in[2];
  const float* wk    = (const float*)d_in[3];
  const float* bk    = (const float*)d_in[4];
  const float* wv    = (const float*)d_in[5];
  const float* bv    = (const float*)d_in[6];
  const float* wo    = (const float*)d_in[7];
  const float* bo    = (const float*)d_in[8];
  const float* gamma = (const float*)d_in[9];
  const float* beta  = (const float*)d_in[10];
  float* outp = (float*)d_out;

  _Float16* WqT = (_Float16*)d_ws;                     // [1024][256]
  _Float16* WkT = WqT + (size_t)INNER * D_MODEL;
  _Float16* WvT = WkT + (size_t)INNER * D_MODEL;
  _Float16* WoT = WvT + (size_t)INNER * D_MODEL;       // [256][1024]

  prep_weights_kernel<<<4096, 256, 0, stream>>>(wq, wk, wv, wo, WqT, WkT, WvT, WoT);

  (void)hipFuncSetAttribute((const void*)attn_fused_kernel,
                            hipFuncAttributeMaxDynamicSharedMemorySize, SMEM_BYTES);

  attn_fused_kernel<<<dim3(B_DIM * N_DIM), dim3(256), SMEM_BYTES, stream>>>(
      x, bq, bk, bv, bo, gamma, beta, WqT, WkT, WvT, WoT, outp);
}
